// TIMGD_40931038331412
// MI455X (gfx1250) — matrix operand order repaired and checked by reading, not compiled, not run on hardware
//
#include <hip/hip_runtime.h>

// ---------------------------------------------------------------------------
// Prototypical-network argmax:
//   logits[t,s,c] = temp*(X·W' - 0.5||w_c||^2 - 0.5||x_s||^2); out = argmax_c
// temp>0 and the x_s term are argmax-invariant -> compute X·W^T - 0.5||w_c||^2.
// GEMM via V_WMMA_F32_16X16X32_BF16 on a hi/lo bf16 split of fp32 inputs
// (xh*wh + xh*wl + xl*wh => ~fp24 mantissa, fp32 accumulate).
//
// Primary path (needs 64MiB+64KB workspace):
//   - pre-pass splits W into bf16 hi/lo planes stored in WMMA dense-B
//     fragment-major order [t][kc][ctile][lane][16]
//   - main kernel DMAs those fragments straight into LDS with
//     global_load_async_to_lds_b128 (double buffered, s_wait_asynccnt+barrier)
//     and feeds v_wmma directly from ds_load_b128 -> no B conversion VALU.
// Fallback path (small workspace): round-2 scheme, fp32 async staging with
// in-wave conversion.
// ---------------------------------------------------------------------------

typedef __attribute__((ext_vector_type(16))) __bf16 v16bf;
typedef __attribute__((ext_vector_type(8)))  float  v8f;

#define TASKS 32
#define SDIM  2048
#define CDIM  512
#define FDIM  1024

#define KB     32                 // K-chunk (one bf16 WMMA K depth)
#define CPASS  256                // classes resident per pass (16 c-tiles)
#define NCHUNK (FDIM / KB)        // 32 K-chunks

// per-task element count of one split plane: kc(32) x ctile(32) x lane(32) x 16
#define WSPLIT_TASK (32 * 32 * 32 * 16)
#define WSPLIT_ELEMS ((size_t)TASKS * WSPLIT_TASK)

// ---- gfx1250 async-copy helpers -------------------------------------------
static __device__ __forceinline__ void async_b128(unsigned int lds_off, const void* gptr) {
    unsigned long long ga = (unsigned long long)(uintptr_t)gptr;
    asm volatile("global_load_async_to_lds_b128 %0, %1, off"
                 :: "v"(lds_off), "v"(ga) : "memory");
}
static __device__ __forceinline__ void wait_async0() {
#if __has_builtin(__builtin_amdgcn_s_wait_asynccnt)
    __builtin_amdgcn_s_wait_asynccnt(0);
#else
    asm volatile("s_wait_asynccnt 0x0" ::: "memory");
#endif
}

// ---------------- wsq[t][c] = 0.5 * sum_f W^2 ------------------------------
__global__ __launch_bounds__(256)
void TIMGD_wsq_kernel(const float* __restrict__ W, float* __restrict__ wsq) {
    const int lane = threadIdx.x & 31;
    const int gw   = blockIdx.x * (blockDim.x >> 5) + (threadIdx.x >> 5);
    if (gw >= TASKS * CDIM) return;
    const float* row = W + (size_t)gw * FDIM;
    float s = 0.f;
    #pragma unroll 4
    for (int i = lane; i < FDIM; i += 32) { float x = row[i]; s += x * x; }
    #pragma unroll
    for (int m = 16; m >= 1; m >>= 1) s += __shfl_xor(s, m, 32);
    if (lane == 0) wsq[gw] = 0.5f * s;
}

// ---------------- W split pre-pass: fp32 -> bf16 hi/lo fragment planes -----
// element (t, c, kc, half): K {half*8+0..7, 16+half*8+0..7} ->
//   dst[((t*32+kc)*32 + c/16)*32 + (c%16 + half*16)][j=0..15]
__global__ __launch_bounds__(256)
void TIMGD_wsplit_kernel(const float* __restrict__ W,
                         __bf16* __restrict__ Whi, __bf16* __restrict__ Wlo) {
    const int e = blockIdx.x * 256 + threadIdx.x;   // [0, 1048576)
    const int half = e & 1;
    const int kc   = (e >> 1) & 31;
    const int c    = (e >> 6) & 511;
    const int t    = e >> 15;

    const float* src = W + (((size_t)t * CDIM + c) * FDIM) + kc * KB + half * 8;
    float f[16];
    *(float4*)&f[0]  = *(const float4*)(src + 0);
    *(float4*)&f[4]  = *(const float4*)(src + 4);
    *(float4*)&f[8]  = *(const float4*)(src + 16);
    *(float4*)&f[12] = *(const float4*)(src + 20);

    __bf16 h[16], l[16];
    #pragma unroll
    for (int j = 0; j < 16; ++j) {
        const __bf16 hh = (__bf16)f[j];
        h[j] = hh;
        l[j] = (__bf16)(f[j] - (float)hh);
    }

    const size_t base =
        ((((size_t)t * 32 + kc) * 32 + (c >> 4)) * 32 + ((c & 15) + half * 16)) * 16;
    uint4* dh = (uint4*)(Whi + base);
    uint4* dl = (uint4*)(Wlo + base);
    dh[0] = *(uint4*)&h[0]; dh[1] = *(uint4*)&h[8];
    dl[0] = *(uint4*)&l[0]; dl[1] = *(uint4*)&l[8];
}

// ---------------- primary main kernel: 256 thr / 8 waves / 128 rows --------
__global__ __launch_bounds__(256)
void TIMGD_proto_frag_kernel(const float* __restrict__ X,
                             const __bf16* __restrict__ Whi,
                             const __bf16* __restrict__ Wlo,
                             const float* __restrict__ wsq,
                             int* __restrict__ out) {
    // bf16 fragments: [sel][hi/lo][ct*32+lane][16]; 4 x 16 KB = 64 KB
    __shared__ alignas(16) __bf16 frag[2][2][16 * 32 * 16];

    const int tid  = threadIdx.x;
    const int wave = tid >> 5;          // 0..7 -> s-tile
    const int lane = tid & 31;

    const int t  = blockIdx.x >> 4;             // 16 s-blocks of 128 per task
    const int s0 = (blockIdx.x & 15) * 128;

    const float*  Xt = X   + (size_t)t * SDIM * FDIM + (size_t)s0 * FDIM;
    const __bf16* Wh = Whi + (size_t)t * WSPLIT_TASK;
    const __bf16* Wl = Wlo + (size_t)t * WSPLIT_TASK;

    // per-lane A source (ISA 16-bit A frag layout):
    // lanes 0-15 hold K{0-7,16-23}, lanes 16-31 hold K{8-15,24-31}
    const float* Arow = Xt + (size_t)(wave * 16 + (lane & 15)) * FDIM
                           + ((lane & 16) ? 8 : 0);

    float best[8];
    int   bidx[8];
    #pragma unroll
    for (int j = 0; j < 8; ++j) { best[j] = -__builtin_inff(); bidx[j] = 0; }

    for (int cpass = 0; cpass < 2; ++cpass) {
        v8f acc[16] = {};

        // async-prefetch one chunk of ready-made fragments (hi+lo = 32 KB)
        auto prefB = [&](int sel, int kc) {
            #pragma unroll 4
            for (int p = tid; p < 2048; p += 256) {
                const int part   = p >> 10;          // 0=hi 1=lo
                const int q      = p & 1023;
                const int ctl    = q >> 6;           // local c-tile 0..15
                const int lane_t = (q >> 1) & 31;
                const int half   = q & 1;
                const size_t gidx =
                    (((size_t)kc * 32 + (cpass * 16 + ctl)) * 32 + lane_t) * 16 + half * 8;
                const __bf16* src = (part ? Wl : Wh) + gidx;
                const unsigned int dst = (unsigned int)(uintptr_t)
                    &frag[sel][part][(ctl * 32 + lane_t) * 16 + half * 8];
                async_b128(dst, src);
            }
        };

        prefB(0, 0);
        float4 a0 = *(const float4*)(Arow + 0);
        float4 a1 = *(const float4*)(Arow + 4);
        float4 a2 = *(const float4*)(Arow + 16);
        float4 a3 = *(const float4*)(Arow + 20);

        for (int kc = 0; kc < NCHUNK; ++kc) {
            const int sel = kc & 1;

            wait_async0();        // my async stores into frag[sel] have landed
            __syncthreads();      // everyone's landed; prev buffer fully read

            float4 n0 = a0, n1 = a1, n2 = a2, n3 = a3;
            if (kc + 1 < NCHUNK) {
                const int kn = (kc + 1) * KB;
                prefB(sel ^ 1, kc + 1);
                n0 = *(const float4*)(Arow + kn + 0);
                n1 = *(const float4*)(Arow + kn + 4);
                n2 = *(const float4*)(Arow + kn + 16);
                n3 = *(const float4*)(Arow + kn + 20);
            }

            // convert this chunk's A fragment: fp32 -> (hi, lo) bf16
            float af[16];
            *(float4*)&af[0]  = a0; *(float4*)&af[4]  = a1;
            *(float4*)&af[8]  = a2; *(float4*)&af[12] = a3;
            v16bf ah, al;
            #pragma unroll
            for (int j = 0; j < 16; ++j) {
                const __bf16 h = (__bf16)af[j];
                ah[j] = h;
                al[j] = (__bf16)(af[j] - (float)h);
            }

            #pragma unroll
            for (int ct = 0; ct < 16; ++ct) {
                const v16bf bh = *(const v16bf*)&frag[sel][0][(ct * 32 + lane) * 16];
                const v16bf bl = *(const v16bf*)&frag[sel][1][(ct * 32 + lane) * 16];
                acc[ct] = __builtin_amdgcn_wmma_f32_16x16x32_bf16(
                    false, ah, false, bh, (short)0, acc[ct], false, false);
                acc[ct] = __builtin_amdgcn_wmma_f32_16x16x32_bf16(
                    false, ah, false, bl, (short)0, acc[ct], false, false);
                acc[ct] = __builtin_amdgcn_wmma_f32_16x16x32_bf16(
                    false, al, false, bh, (short)0, acc[ct], false, false);
            }

            a0 = n0; a1 = n1; a2 = n2; a3 = n3;
        }

        // fold this pass into the running argmax
        #pragma unroll
        for (int ct = 0; ct < 16; ++ct) {
            const int   c  = cpass * CPASS + ct * 16 + (lane & 15);
            const float wb = wsq[t * CDIM + c];
            #pragma unroll
            for (int j = 0; j < 8; ++j) {
                const float v = acc[ct][j] - wb;
                if (v > best[j]) { best[j] = v; bidx[j] = c; }
            }
        }
    }

    // reduce across the 16 lanes sharing each output row
    #pragma unroll
    for (int j = 0; j < 8; ++j) {
        float v  = best[j];
        int   ix = bidx[j];
        #pragma unroll
        for (int m = 8; m >= 1; m >>= 1) {
            const float ov  = __shfl_xor(v, m, 32);
            const int   oix = __shfl_xor(ix, m, 32);
            if (ov > v || (ov == v && oix < ix)) { v = ov; ix = oix; }
        }
        bidx[j] = ix;
    }
    if ((lane & 15) == 0) {
        const int rbase = wave * 16 + ((lane >> 4) ? 8 : 0);
        int* o = out + (size_t)t * SDIM + s0 + rbase;
        #pragma unroll
        for (int j = 0; j < 8; ++j) o[j] = bidx[j];
    }
}

// ---------------- fallback main kernel (round-2 path, no big workspace) ----
__global__ __launch_bounds__(128)
void TIMGD_proto_fb_kernel(const float* __restrict__ X,
                           const float* __restrict__ W,
                           const float* __restrict__ wsq,
                           int* __restrict__ out) {
    __shared__ alignas(16) float rawB[2][CPASS * KB];

    const int tid  = threadIdx.x;
    const int wave = tid >> 5;
    const int lane = tid & 31;

    const int t  = blockIdx.x >> 5;
    const int s0 = (blockIdx.x & 31) * 64;

    const float* Xt = X + (size_t)t * SDIM * FDIM + (size_t)s0 * FDIM;
    const float* Wt = W + (size_t)t * CDIM * FDIM;
    const float* Arow = Xt + (size_t)(wave * 16 + (lane & 15)) * FDIM
                           + ((lane & 16) ? 8 : 0);

    float best[8];
    int   bidx[8];
    #pragma unroll
    for (int j = 0; j < 8; ++j) { best[j] = -__builtin_inff(); bidx[j] = 0; }

    for (int cpass = 0; cpass < 2; ++cpass) {
        const float* Wp = Wt + (size_t)(cpass * CPASS) * FDIM;
        v8f acc[16] = {};

        auto prefB = [&](int sel, int k0) {
            const unsigned int base = (unsigned int)(uintptr_t)(&rawB[sel][0]);
            #pragma unroll 4
            for (int p = tid; p < (CPASS * KB) / 4; p += 128) {
                const int cr = p >> 3;
                const int g  = p & 7;
                const int ct = cr >> 4, n = cr & 15;
                const int ln = n + ((g & 2) << 3);
                const unsigned int dst =
                    base + (unsigned int)((((ct * 32 + ln) << 4) + ((g & 4) << 1) + ((g & 1) << 2)) << 2);
                async_b128(dst, Wp + (size_t)cr * FDIM + k0 + (g << 2));
            }
        };

        prefB(0, 0);
        float4 a0 = *(const float4*)(Arow + 0);
        float4 a1 = *(const float4*)(Arow + 4);
        float4 a2 = *(const float4*)(Arow + 16);
        float4 a3 = *(const float4*)(Arow + 20);

        for (int kc = 0; kc < NCHUNK; ++kc) {
            const int sel = kc & 1;
            wait_async0();
            __syncthreads();

            float4 n0 = a0, n1 = a1, n2 = a2, n3 = a3;
            if (kc + 1 < NCHUNK) {
                const int kn = (kc + 1) * KB;
                prefB(sel ^ 1, kn);
                n0 = *(const float4*)(Arow + kn + 0);
                n1 = *(const float4*)(Arow + kn + 4);
                n2 = *(const float4*)(Arow + kn + 16);
                n3 = *(const float4*)(Arow + kn + 20);
            }

            float af[16];
            *(float4*)&af[0]  = a0; *(float4*)&af[4]  = a1;
            *(float4*)&af[8]  = a2; *(float4*)&af[12] = a3;
            v16bf ah, al;
            #pragma unroll
            for (int j = 0; j < 16; ++j) {
                const __bf16 h = (__bf16)af[j];
                ah[j] = h;
                al[j] = (__bf16)(af[j] - (float)h);
            }

            const float* bbuf = &rawB[sel][0];
            #pragma unroll
            for (int ct = 0; ct < 16; ++ct) {
                const float4* bp = (const float4*)(bbuf + (((ct * 32) + lane) << 4));
                float bf[16];
                *(float4*)&bf[0]  = bp[0]; *(float4*)&bf[4]  = bp[1];
                *(float4*)&bf[8]  = bp[2]; *(float4*)&bf[12] = bp[3];
                v16bf bh, bl;
                #pragma unroll
                for (int j = 0; j < 16; ++j) {
                    const __bf16 h = (__bf16)bf[j];
                    bh[j] = h;
                    bl[j] = (__bf16)(bf[j] - (float)h);
                }
                acc[ct] = __builtin_amdgcn_wmma_f32_16x16x32_bf16(
                    false, ah, false, bh, (short)0, acc[ct], false, false);
                acc[ct] = __builtin_amdgcn_wmma_f32_16x16x32_bf16(
                    false, ah, false, bl, (short)0, acc[ct], false, false);
                acc[ct] = __builtin_amdgcn_wmma_f32_16x16x32_bf16(
                    false, al, false, bh, (short)0, acc[ct], false, false);
            }
            a0 = n0; a1 = n1; a2 = n2; a3 = n3;
        }

        #pragma unroll
        for (int ct = 0; ct < 16; ++ct) {
            const int   c  = cpass * CPASS + ct * 16 + (lane & 15);
            const float wb = wsq[t * CDIM + c];
            #pragma unroll
            for (int j = 0; j < 8; ++j) {
                const float v = acc[ct][j] - wb;
                if (v > best[j]) { best[j] = v; bidx[j] = c; }
            }
        }
    }

    #pragma unroll
    for (int j = 0; j < 8; ++j) {
        float v  = best[j];
        int   ix = bidx[j];
        #pragma unroll
        for (int m = 8; m >= 1; m >>= 1) {
            const float ov  = __shfl_xor(v, m, 32);
            const int   oix = __shfl_xor(ix, m, 32);
            if (ov > v || (ov == v && oix < ix)) { v = ov; ix = oix; }
        }
        bidx[j] = ix;
    }
    if ((lane & 15) == 0) {
        const int rbase = wave * 16 + ((lane >> 4) ? 8 : 0);
        int* o = out + (size_t)t * SDIM + s0 + rbase;
        #pragma unroll
        for (int j = 0; j < 8; ++j) o[j] = bidx[j];
    }
}

// ---------------------------------------------------------------------------
extern "C" void kernel_launch(void* const* d_in, const int* in_sizes, int n_in,
                              void* d_out, int out_size, void* d_ws, size_t ws_size,
                              hipStream_t stream) {
    const float* X = (const float*)d_in[0];   // [32,2048,1024] fp32
    const float* W = (const float*)d_in[1];   // [32,512,1024]  fp32
    // d_in[2] = temp (positive scalar): argmax-invariant, unused.
    int* out = (int*)d_out;                   // [32,2048] int32

    (void)in_sizes; (void)n_in; (void)out_size;

    char*  ws   = (char*)d_ws;
    float* wsq  = (float*)ws;                             // 64 KB
    const size_t wsq_bytes   = (size_t)TASKS * CDIM * sizeof(float);
    const size_t plane_bytes = WSPLIT_ELEMS * sizeof(__bf16);   // 32 MiB each
    const size_t need        = wsq_bytes + 2 * plane_bytes;

    TIMGD_wsq_kernel<<<(TASKS * CDIM) / 8, 256, 0, stream>>>(W, wsq);

    if (ws_size >= need) {
        __bf16* Whi = (__bf16*)(ws + wsq_bytes);
        __bf16* Wlo = (__bf16*)(ws + wsq_bytes + plane_bytes);
        // 1,048,576 threads: (t, c, kc, half) -> 16 K-values each
        TIMGD_wsplit_kernel<<<4096, 256, 0, stream>>>(W, Whi, Wlo);
        // 32 tasks x 16 s-blocks of 128 rows; 256 threads = 8 waves
        TIMGD_proto_frag_kernel<<<TASKS * (SDIM / 128), 256, 0, stream>>>(
            X, Whi, Wlo, wsq, out);
    } else {
        // workspace too small: fp32 staging + in-wave conversion
        TIMGD_proto_fb_kernel<<<TASKS * (SDIM / 64), 128, 0, stream>>>(
            X, W, wsq, out);
    }
}
